// NeuronMixtralDecoderLayer_20486994002048
// MI455X (gfx1250) — compile-verified, matrix operand order checked by reading, not run on hardware
//
#include <hip/hip_runtime.h>
#include <hip/hip_bf16.h>

// ---------------------------------------------------------------------------
// Types for CDNA5 WMMA (wave32)
// ---------------------------------------------------------------------------
typedef __bf16 bf16_t;
typedef __attribute__((ext_vector_type(16))) __bf16 v16bf;
typedef __attribute__((ext_vector_type(8)))  __bf16 v8bf;
typedef __attribute__((ext_vector_type(8)))  float  v8f;

#define D_   2048
#define S_   2048
#define N_   2048   // B*S tokens
#define H_   32
#define KH_  8
#define HD_  64
#define E_   8
#define F_   7168

__device__ __forceinline__ bf16_t f2bf(float f) {
    union { float f; unsigned u; } v; v.f = f;
    unsigned r = (v.u + 0x7FFFu + ((v.u >> 16) & 1u)) >> 16;
    union { unsigned short s; bf16_t b; } o; o.s = (unsigned short)r;
    return o.b;
}
__device__ __forceinline__ float bf2f(bf16_t b) {
    union { unsigned short s; bf16_t b; } i; i.b = b;
    union { unsigned u; float f; } o; o.u = ((unsigned)i.s) << 16;
    return o.f;
}
__device__ __forceinline__ unsigned pack2bf(float lo, float hi) {
    union { bf16_t b; unsigned short s; } a, c;
    a.b = f2bf(lo); c.b = f2bf(hi);
    return ((unsigned)c.s << 16) | (unsigned)a.s;
}
// Assemble a 16x16x32 operand fragment from two 16B LDS chunks.
// Lane K-runs: [o,o+8) U [16+o,16+o+8), o = 8*(lane>=16). `p` points at the
// 32-half row start (16B aligned); chunks hi and 2+hi are the two runs.
__device__ __forceinline__ v16bf frag_from(const v8bf* p, int hi) {
    v8bf t0 = p[hi], t1 = p[2 + hi];
    v16bf f;
    #pragma unroll
    for (int q = 0; q < 8; ++q) { f[q] = t0[q]; f[8 + q] = t1[q]; }
    return f;
}

// ---------------------------------------------------------------------------
// WMMA GEMM: C[M,N] = A[M,K](bf16,row-major) * B[K,N](f32,row-major)
// Block 256 thr = 8 waves; tile 128x128, K-step 32; wave tile 32x64 = 8 WMMA.
// A tile staged with GLOBAL_LOAD_ASYNC_TO_LDS_B128 (ASYNCcnt); B converted
// f32->bf16 in flight into a transposed [n][k] LDS tile with packed b32 stores.
// Fused epilogue: optional elementwise add (residual), optional per-row scale
// (MoE combine weight), optional accumulate into C.
// ---------------------------------------------------------------------------
#define BM 128
#define BN 128
#define BK 32
#define LDSP 40   // padded LDS row (80B: rows 16B-aligned, bank-staggered)

__global__ __launch_bounds__(256)
void wmma_gemm_kernel(const bf16_t* __restrict__ A, const float* __restrict__ B,
                      float* __restrict__ C, int M, int Nn, int K,
                      int lda, int ldb, int ldc,
                      const float* __restrict__ addsrc,   // nullable
                      const float* __restrict__ rowscale, // nullable
                      int rsStride, int accum)
{
    __shared__ bf16_t As[BM][LDSP];   // [m][k]
    __shared__ bf16_t Bs[BN][LDSP];   // [n][k] (transposed at store)

    const int tid  = threadIdx.x;
    const int m0   = blockIdx.y * BM;
    const int n0   = blockIdx.x * BN;
    const int wave = tid >> 5;
    const int lane = tid & 31;
    const int wy   = wave >> 1;       // 0..3 -> 32 rows
    const int wx   = wave & 1;        // 0..1 -> 64 cols
    const int l16  = lane & 15;
    const int hi   = lane >> 4;

    v8f acc[2][4];
    #pragma unroll
    for (int i = 0; i < 2; ++i)
        #pragma unroll
        for (int j = 0; j < 4; ++j)
            #pragma unroll
            for (int r = 0; r < 8; ++r) acc[i][j][r] = 0.0f;

    for (int k0 = 0; k0 < K; k0 += BK) {
        // ---- A tile: 128x32 bf16; 2 async b128 per thread ----
        #pragma unroll
        for (int c = 0; c < 2; ++c) {
            int lin = tid + c * 256;
            int r   = lin >> 2;
            int kk  = (lin & 3) << 3;
            unsigned lds_addr = (unsigned)(uintptr_t)&As[r][kk];
            unsigned long long gaddr =
                (unsigned long long)(uintptr_t)(A + (size_t)(m0 + r) * lda + k0 + kk);
            asm volatile("global_load_async_to_lds_b128 %0, %1, off"
                         :: "v"(lds_addr), "v"(gaddr) : "memory");
        }
        // ---- B tile: 32x128 f32 -> bf16 transposed [n][k] ----
        #pragma unroll
        for (int c = 0; c < 2; ++c) {
            int lin = tid + c * 256;
            int nn4 = lin & 31;
            int kk2 = lin >> 5;
            const float* bp = B + (size_t)(k0 + 2 * kk2) * ldb + n0 + 4 * nn4;
            float4 r0 = *(const float4*)bp;
            float4 r1 = *(const float4*)(bp + ldb);
            const float* p0 = (const float*)&r0;
            const float* p1 = (const float*)&r1;
            #pragma unroll
            for (int i = 0; i < 4; ++i)
                *(unsigned*)&Bs[4 * nn4 + i][2 * kk2] = pack2bf(p0[i], p1[i]);
        }
        asm volatile("s_wait_asynccnt 0x0" ::: "memory");
        __syncthreads();

        v16bf a[2], b[4];
        #pragma unroll
        for (int i = 0; i < 2; ++i)
            a[i] = frag_from((const v8bf*)&As[wy * 32 + i * 16 + l16][0], hi);
        #pragma unroll
        for (int j = 0; j < 4; ++j)
            b[j] = frag_from((const v8bf*)&Bs[wx * 64 + j * 16 + l16][0], hi);
        #pragma unroll
        for (int i = 0; i < 2; ++i)
            #pragma unroll
            for (int j = 0; j < 4; ++j)
                acc[i][j] = __builtin_amdgcn_wmma_f32_16x16x32_bf16(
                    false, a[i], false, b[j], (short)0, acc[i][j], false, false);
        __syncthreads();
    }

    // ---- fused epilogue + store ----
    #pragma unroll
    for (int i = 0; i < 2; ++i)
        #pragma unroll
        for (int j = 0; j < 4; ++j) {
            int rbase = m0 + wy * 32 + i * 16 + hi * 8;
            int cbase = n0 + wx * 64 + j * 16 + l16;
            #pragma unroll
            for (int r = 0; r < 8; ++r) {
                size_t idx = (size_t)(rbase + r) * ldc + cbase;
                float vv = acc[i][j][r];
                if (rowscale) vv *= rowscale[(size_t)(rbase + r) * rsStride];
                if (addsrc)   vv += addsrc[idx];
                if (accum)    vv += C[idx];
                C[idx] = vv;
            }
        }
}

// ---------------------------------------------------------------------------
// WMMA flash attention (causal, GQA). Block = 128 queries x 1 head, 8 waves
// x 16 queries. S = Q K^T and O += P V both on v_wmma_f32_16x16x32_bf16.
// Online softmax operates directly on C-layout accumulators; row reductions
// via __shfl_xor within each 16-lane half (one row per half per VGPR).
// P transposed C-layout -> A-layout through a per-wave-private LDS region.
// ---------------------------------------------------------------------------
#define AQ  128
#define AKT 64
#define APAD 72   // padded row in halves (144B, 16B-aligned)

__global__ __launch_bounds__(256)
void attn_wmma_kernel(const float* __restrict__ q, const float* __restrict__ k,
                      const float* __restrict__ v, bf16_t* __restrict__ o)
{
    __shared__ bf16_t Qs[AQ][APAD];    // [q][d]
    __shared__ bf16_t Ks[AKT][APAD];   // [key][d]  (B-layout for QK^T: N=key,K=d)
    __shared__ bf16_t Vs[HD_][APAD];   // [d][key]  (B-layout for PV:   N=d,K=key)
    __shared__ bf16_t Ps[AQ][APAD];    // [q][key]  (A-layout source for PV)

    const int h    = blockIdx.y;
    const int qt0  = blockIdx.x * AQ;
    const int tid  = threadIdx.x;
    const int wave = tid >> 5;
    const int lane = tid & 31;
    const int l16  = lane & 15;
    const int hi   = lane >> 4;
    const int kh   = h >> 2;               // GQA: 4 q-heads per kv head
    const int qb   = qt0 + wave * 16;      // this wave's query base

    // ---- stage Q tile: 128x64 f32 -> bf16 ----
    #pragma unroll
    for (int c = 0; c < 4; ++c) {
        int lin = tid + c * 256;           // 0..1023 (8-elem units)
        int r   = lin >> 3;                // 0..127
        int dd  = (lin & 7) * 8;           // 0..56
        const float* src = q + (size_t)(qt0 + r) * (H_ * HD_) + h * HD_ + dd;
        float4 a0 = *(const float4*)src;
        float4 a1 = *(const float4*)(src + 4);
        unsigned* dst = (unsigned*)&Qs[r][dd];
        dst[0] = pack2bf(a0.x, a0.y); dst[1] = pack2bf(a0.z, a0.w);
        dst[2] = pack2bf(a1.x, a1.y); dst[3] = pack2bf(a1.z, a1.w);
    }

    float mrow[8], lrow[8];
    #pragma unroll
    for (int r = 0; r < 8; ++r) { mrow[r] = -1e30f; lrow[r] = 0.f; }
    v8f oacc[4];
    #pragma unroll
    for (int j = 0; j < 4; ++j)
        #pragma unroll
        for (int r = 0; r < 8; ++r) oacc[j][r] = 0.f;

    for (int kt0 = 0; kt0 <= qt0 + AQ - 1; kt0 += AKT) {
        __syncthreads();   // prev-iter reads done (and initial Qs staged)
        // ---- stage K tile [key][d] ----
        #pragma unroll
        for (int c = 0; c < 2; ++c) {
            int lin = tid + c * 256;       // 0..511
            int r   = lin >> 3;            // key 0..63
            int dd  = (lin & 7) * 8;
            const float* src = k + (size_t)(kt0 + r) * (KH_ * HD_) + kh * HD_ + dd;
            float4 a0 = *(const float4*)src;
            float4 a1 = *(const float4*)(src + 4);
            unsigned* dst = (unsigned*)&Ks[r][dd];
            dst[0] = pack2bf(a0.x, a0.y); dst[1] = pack2bf(a0.z, a0.w);
            dst[2] = pack2bf(a1.x, a1.y); dst[3] = pack2bf(a1.z, a1.w);
        }
        // ---- stage V transposed [d][key] (pack key pairs) ----
        #pragma unroll
        for (int c = 0; c < 2; ++c) {
            int lin = tid + c * 256;       // 0..511
            int d4  = lin & 15;            // d quad 0..15
            int j2  = lin >> 4;            // key pair 0..31
            const float* vp = v + (size_t)(kt0 + 2 * j2) * (KH_ * HD_) + kh * HD_ + 4 * d4;
            float4 r0 = *(const float4*)vp;
            float4 r1 = *(const float4*)(vp + KH_ * HD_);
            const float* p0 = (const float*)&r0;
            const float* p1 = (const float*)&r1;
            #pragma unroll
            for (int i = 0; i < 4; ++i)
                *(unsigned*)&Vs[4 * d4 + i][2 * j2] = pack2bf(p0[i], p1[i]);
        }
        __syncthreads();

        if (kt0 <= qb + 15) {              // wave-uniform causal tile skip
            // ---- S = Q K^T ----
            v8f sacc[4];
            #pragma unroll
            for (int j = 0; j < 4; ++j)
                #pragma unroll
                for (int r = 0; r < 8; ++r) sacc[j][r] = 0.f;
            #pragma unroll
            for (int ks = 0; ks < 2; ++ks) {
                v16bf af = frag_from((const v8bf*)&Qs[wave * 16 + l16][ks * 32], hi);
                #pragma unroll
                for (int j = 0; j < 4; ++j) {
                    v16bf bf_ = frag_from((const v8bf*)&Ks[j * 16 + l16][ks * 32], hi);
                    sacc[j] = __builtin_amdgcn_wmma_f32_16x16x32_bf16(
                        false, af, false, bf_, (short)0, sacc[j], false, false);
                }
            }
            // ---- online softmax on C-layout accumulators ----
            #pragma unroll
            for (int r = 0; r < 8; ++r) {
                int qrow = qb + r + 8 * hi;
                float tm = -1e30f;
                #pragma unroll
                for (int j = 0; j < 4; ++j) {
                    float s = sacc[j][r] * 0.125f;          // 1/sqrt(64)
                    int key = kt0 + j * 16 + l16;
                    if (key > qrow) s = -1e30f;             // causal mask
                    sacc[j][r] = s;
                    tm = fmaxf(tm, s);
                }
                tm = fmaxf(tm, __shfl_xor(tm, 1));
                tm = fmaxf(tm, __shfl_xor(tm, 2));
                tm = fmaxf(tm, __shfl_xor(tm, 4));
                tm = fmaxf(tm, __shfl_xor(tm, 8));
                float nm = fmaxf(mrow[r], tm);
                float sc = __expf(mrow[r] - nm);
                mrow[r] = nm;
                lrow[r] *= sc;
                #pragma unroll
                for (int j = 0; j < 4; ++j) oacc[j][r] *= sc;
                float rs = 0.f;
                #pragma unroll
                for (int j = 0; j < 4; ++j) {
                    float p = __expf(sacc[j][r] - nm);
                    sacc[j][r] = p;
                    rs += p;
                }
                rs += __shfl_xor(rs, 1);
                rs += __shfl_xor(rs, 2);
                rs += __shfl_xor(rs, 4);
                rs += __shfl_xor(rs, 8);
                lrow[r] += rs;
            }
            // ---- P: C-layout -> A-layout via per-wave-private LDS ----
            #pragma unroll
            for (int r = 0; r < 8; ++r)
                #pragma unroll
                for (int j = 0; j < 4; ++j)
                    Ps[wave * 16 + r + 8 * hi][j * 16 + l16] = f2bf(sacc[j][r]);
            // same-wave DS ops are in-order: no barrier needed
            // ---- O += P V ----
            #pragma unroll
            for (int ks = 0; ks < 2; ++ks) {
                v16bf af = frag_from((const v8bf*)&Ps[wave * 16 + l16][ks * 32], hi);
                #pragma unroll
                for (int j = 0; j < 4; ++j) {
                    v16bf bf_ = frag_from((const v8bf*)&Vs[j * 16 + l16][ks * 32], hi);
                    oacc[j] = __builtin_amdgcn_wmma_f32_16x16x32_bf16(
                        false, af, false, bf_, (short)0, oacc[j], false, false);
                }
            }
        }
    }
    // ---- O /= l, write bf16 ----
    #pragma unroll
    for (int j = 0; j < 4; ++j)
        #pragma unroll
        for (int r = 0; r < 8; ++r) {
            float val = oacc[j][r] / lrow[r];
            o[(size_t)(qb + r + 8 * hi) * (H_ * HD_) + h * HD_ + j * 16 + l16] =
                f2bf(val);
        }
}

// ---------------------------------------------------------------------------
// RMSNorm (fp32 in) -> bf16 out
// ---------------------------------------------------------------------------
__global__ __launch_bounds__(256)
void rmsnorm_kernel(const float* __restrict__ x, const float* __restrict__ w,
                    bf16_t* __restrict__ o)
{
    __shared__ float red[256];
    int row = blockIdx.x, tid = threadIdx.x;
    const float* xr = x + (size_t)row * D_;
    float s = 0.f;
    for (int d = tid; d < D_; d += 256) { float t = xr[d]; s += t * t; }
    red[tid] = s; __syncthreads();
    for (int st = 128; st > 0; st >>= 1) {
        if (tid < st) red[tid] += red[tid + st];
        __syncthreads();
    }
    float inv = rsqrtf(red[0] / (float)D_ + 1e-5f);
    for (int d = tid; d < D_; d += 256)
        o[(size_t)row * D_ + d] = f2bf(xr[d] * inv * w[d]);
}

// ---------------------------------------------------------------------------
// RoPE applied in place to q [S,H*HD] and k [S,KH*HD]
// ---------------------------------------------------------------------------
__global__ __launch_bounds__(256)
void rope_kernel(float* __restrict__ q, float* __restrict__ k,
                 const int* __restrict__ pos)
{
    int g = blockIdx.x * blockDim.x + threadIdx.x;
    int p    = g & 31;
    int slot = (g >> 5) % (H_ + KH_);
    int s    = g / (32 * (H_ + KH_));
    if (s >= S_) return;
    float* base = (slot < H_)
        ? q + (size_t)s * (H_ * HD_) + slot * HD_
        : k + (size_t)s * (KH_ * HD_) + (slot - H_) * HD_;
    float posf = (float)pos[s];
    float freq = __expf(-((2.0f * (float)p) / 64.0f) * 13.815510557964274f);
    float ang = posf * freq;
    float c = __cosf(ang), sn = __sinf(ang);
    float x1 = base[p], x2 = base[p + 32];
    base[p]      = x1 * c - x2 * sn;
    base[p + 32] = x2 * c + x1 * sn;
}

// ---------------------------------------------------------------------------
// Router: softmax over 8 experts, top-2, renormalize -> dense combine[N,E]
// ---------------------------------------------------------------------------
__global__ __launch_bounds__(256)
void router_kernel(const bf16_t* __restrict__ hb, const float* __restrict__ Wr,
                   float* __restrict__ combine)
{
    int n = blockIdx.x * blockDim.x + threadIdx.x;
    if (n >= N_) return;
    float logit[E_];
    #pragma unroll
    for (int e = 0; e < E_; ++e) logit[e] = 0.f;
    const bf16_t* xr = hb + (size_t)n * D_;
    for (int d = 0; d < D_; ++d) {
        float xv = bf2f(xr[d]);
        #pragma unroll
        for (int e = 0; e < E_; ++e) logit[e] += xv * Wr[d * E_ + e];
    }
    float mx = logit[0];
    #pragma unroll
    for (int e = 1; e < E_; ++e) mx = fmaxf(mx, logit[e]);
    float p[E_], sum = 0.f;
    #pragma unroll
    for (int e = 0; e < E_; ++e) { p[e] = __expf(logit[e] - mx); sum += p[e]; }
    #pragma unroll
    for (int e = 0; e < E_; ++e) p[e] /= sum;
    int i1 = 0;
    #pragma unroll
    for (int e = 1; e < E_; ++e) if (p[e] > p[i1]) i1 = e;
    int i2 = (i1 == 0) ? 1 : 0;
    #pragma unroll
    for (int e = 0; e < E_; ++e) { if (e != i1 && p[e] > p[i2]) i2 = e; }
    float tot = p[i1] + p[i2];
    #pragma unroll
    for (int e = 0; e < E_; ++e) combine[n * E_ + e] = 0.f;
    combine[n * E_ + i1] = p[i1] / tot;
    combine[n * E_ + i2] = p[i2] / tot;
}

// ---------------------------------------------------------------------------
// SiLU(gate)*up -> bf16
// ---------------------------------------------------------------------------
__global__ __launch_bounds__(256)
void silu_mul_kernel(const float* __restrict__ g, const float* __restrict__ u,
                     bf16_t* __restrict__ o, int n)
{
    int i = blockIdx.x * blockDim.x + threadIdx.x;
    if (i < n) {
        float gv = g[i];
        float s = gv / (1.0f + __expf(-gv));
        o[i] = f2bf(s * u[i]);
    }
}

// ---------------------------------------------------------------------------
// Orchestration
// ---------------------------------------------------------------------------
extern "C" void kernel_launch(void* const* d_in, const int* in_sizes, int n_in,
                              void* d_out, int out_size, void* d_ws, size_t ws_size,
                              hipStream_t stream) {
    const float* hidden = (const float*)d_in[0];
    const int*   pos    = (const int*)d_in[1];
    const float* ln1_w  = (const float*)d_in[2];
    const float* Wq     = (const float*)d_in[3];
    const float* Wk     = (const float*)d_in[4];
    const float* Wv     = (const float*)d_in[5];
    const float* Wo     = (const float*)d_in[6];
    const float* ln2_w  = (const float*)d_in[7];
    const float* Wr     = (const float*)d_in[8];
    const float* W1     = (const float*)d_in[9];
    const float* W3     = (const float*)d_in[10];
    const float* W2     = (const float*)d_in[11];
    float* out = (float*)d_out;

    char* w = (char*)d_ws;
    size_t off = 0;
    auto alloc = [&](size_t bytes) -> void* {
        void* p = w + off;
        off += (bytes + 255) & ~(size_t)255;
        return p;
    };
    bf16_t* hbf    = (bf16_t*)alloc((size_t)N_ * D_ * 2);
    bf16_t* hbf2   = (bf16_t*)alloc((size_t)N_ * D_ * 2);
    bf16_t* attnbf = (bf16_t*)alloc((size_t)N_ * D_ * 2);
    float*  qbuf   = (float*)alloc((size_t)N_ * (H_ * HD_) * 4);
    float*  kbuf   = (float*)alloc((size_t)N_ * (KH_ * HD_) * 4);
    float*  vbuf   = (float*)alloc((size_t)N_ * (KH_ * HD_) * 4);
    float*  comb   = (float*)alloc((size_t)N_ * E_ * 4);
    float*  gate   = (float*)alloc((size_t)N_ * F_ * 4);
    float*  up     = (float*)alloc((size_t)N_ * F_ * 4);
    bf16_t* actbf  = (bf16_t*)alloc((size_t)N_ * F_ * 2);

    const int NF = N_ * F_;

    // 1. RMSNorm 1
    rmsnorm_kernel<<<N_, 256, 0, stream>>>(hidden, ln1_w, hbf);

    // 2-4. Q/K/V projections (WMMA)
    wmma_gemm_kernel<<<dim3((H_ * HD_) / BN, N_ / BM), 256, 0, stream>>>(
        hbf, Wq, qbuf, N_, H_ * HD_, D_, D_, H_ * HD_, H_ * HD_,
        nullptr, nullptr, 0, 0);
    wmma_gemm_kernel<<<dim3((KH_ * HD_) / BN, N_ / BM), 256, 0, stream>>>(
        hbf, Wk, kbuf, N_, KH_ * HD_, D_, D_, KH_ * HD_, KH_ * HD_,
        nullptr, nullptr, 0, 0);
    wmma_gemm_kernel<<<dim3((KH_ * HD_) / BN, N_ / BM), 256, 0, stream>>>(
        hbf, Wv, vbuf, N_, KH_ * HD_, D_, D_, KH_ * HD_, KH_ * HD_,
        nullptr, nullptr, 0, 0);

    // 5. RoPE
    {
        int total = S_ * (H_ + KH_) * 32;
        rope_kernel<<<(total + 255) / 256, 256, 0, stream>>>(qbuf, kbuf, pos);
    }

    // 6. WMMA flash attention -> bf16
    attn_wmma_kernel<<<dim3(S_ / AQ, H_), 256, 0, stream>>>(qbuf, kbuf, vbuf, attnbf);

    // 7. Output projection + fused residual add -> d_out
    wmma_gemm_kernel<<<dim3(D_ / BN, N_ / BM), 256, 0, stream>>>(
        attnbf, Wo, out, N_, D_, H_ * HD_, H_ * HD_, D_, D_,
        hidden, nullptr, 0, 0);

    // 8. RMSNorm 2
    rmsnorm_kernel<<<N_, 256, 0, stream>>>(out, ln2_w, hbf2);

    // 9. Router
    router_kernel<<<(N_ + 255) / 256, 256, 0, stream>>>(hbf2, Wr, comb);

    // 10. Experts (dense-all-experts * combine == reference semantics);
    //     down-proj fuses combine-scale + accumulate into d_out.
    for (int e = 0; e < E_; ++e) {
        const float* W1e = W1 + (size_t)e * D_ * F_;
        const float* W3e = W3 + (size_t)e * D_ * F_;
        const float* W2e = W2 + (size_t)e * F_ * D_;
        wmma_gemm_kernel<<<dim3(F_ / BN, N_ / BM), 256, 0, stream>>>(
            hbf2, W1e, gate, N_, F_, D_, D_, F_, F_,
            nullptr, nullptr, 0, 0);
        wmma_gemm_kernel<<<dim3(F_ / BN, N_ / BM), 256, 0, stream>>>(
            hbf2, W3e, up, N_, F_, D_, D_, F_, F_,
            nullptr, nullptr, 0, 0);
        silu_mul_kernel<<<(NF + 255) / 256, 256, 0, stream>>>(gate, up, actbf, NF);
        wmma_gemm_kernel<<<dim3(D_ / BN, N_ / BM), 256, 0, stream>>>(
            actbf, W2e, out, N_, D_, F_, F_, D_, D_,
            nullptr, comb + e, E_, 1);
    }
    (void)in_sizes; (void)n_in; (void)out_size; (void)ws_size;
}
